// INT8QuantizedLinear_76596446757272
// MI455X (gfx1250) — 8-bit matrix operand order repaired and checked by reading, compiled, not run on hardware
//
#include <hip/hip_runtime.h>
#include <hip/hip_bf16.h>
#include <stdint.h>

typedef int   v8i __attribute__((ext_vector_type(8)));
typedef int   v4i __attribute__((ext_vector_type(4)));
typedef int   v2i __attribute__((ext_vector_type(2)));
typedef float v4f __attribute__((ext_vector_type(4)));

#define M_TOK 8192
#define K_DIM 1024
#define N_DIM 4096

__global__ __launch_bounds__(256) void pack_w_kernel(const int* __restrict__ w32,
                                                     uint32_t* __restrict__ w8) {
    size_t i = (size_t)blockIdx.x * 256 + threadIdx.x;
    v4i w = *reinterpret_cast<const v4i*>(w32 + i * 4);
    uint32_t p = (uint32_t)(w[0] & 0xFF)
               | ((uint32_t)(w[1] & 0xFF) << 8)
               | ((uint32_t)(w[2] & 0xFF) << 16)
               | ((uint32_t)(w[3] & 0xFF) << 24);
    w8[i] = p;
}

__global__ __launch_bounds__(256) void quant_kernel(const float* __restrict__ x,
                                                    uint32_t* __restrict__ xq,
                                                    float* __restrict__ iscale) {
    const int lane = threadIdx.x & 31;
    const int wave = threadIdx.x >> 5;
    const int m    = blockIdx.x * 8 + wave;

    const v4f* xrow = reinterpret_cast<const v4f*>(x + (size_t)m * K_DIM);
    v4f vals[8];
    float amax = 0.f;
#pragma unroll
    for (int i = 0; i < 8; ++i) {
        v4f v = xrow[i * 32 + lane];
        vals[i] = v;
        amax = fmaxf(amax, fmaxf(fmaxf(fabsf(v.x), fabsf(v.y)),
                                 fmaxf(fabsf(v.z), fabsf(v.w))));
    }
#pragma unroll
    for (int off = 16; off > 0; off >>= 1)
        amax = fmaxf(amax, __shfl_xor(amax, off, 32));

    const float scale = (amax == 0.f) ? 1.f : amax * (1.f / 127.f);
    const float r     = 1.f / scale;

    uint32_t* qrow = xq + (size_t)m * (K_DIM / 4);
#pragma unroll
    for (int i = 0; i < 8; ++i) {
        v4f v = vals[i];
        int q0 = max(-128, min(127, __float2int_rn(v.x * r)));
        int q1 = max(-128, min(127, __float2int_rn(v.y * r)));
        int q2 = max(-128, min(127, __float2int_rn(v.z * r)));
        int q3 = max(-128, min(127, __float2int_rn(v.w * r)));
        uint32_t p = (uint32_t)(q0 & 0xFF)
                   | ((uint32_t)(q1 & 0xFF) << 8)
                   | ((uint32_t)(q2 & 0xFF) << 16)
                   | ((uint32_t)(q3 & 0xFF) << 24);
        qrow[i * 32 + lane] = p;
    }
    if (lane == 0) iscale[m] = scale;
}

__global__ __launch_bounds__(256) void gemm_kernel(const int8_t* __restrict__ xq,
                                                   const float* __restrict__ iscale,
                                                   const int8_t* __restrict__ wq,
                                                   const float* __restrict__ wscale,
                                                   const float* __restrict__ bias,
                                                   float* __restrict__ out) {
    const int lane = threadIdx.x & 31;
    const int wave = threadIdx.x >> 5;
    const int hi   = lane >> 4;
    const int ln   = lane & 15;

    const int n0 = blockIdx.x * 128;
    const int m0 = blockIdx.y * 128 + wave * 16;

    const int8_t* arow = xq + (size_t)(m0 + ln) * K_DIM + hi * 16;

    v8i acc[8] = {};

    for (int kt = 0; kt < K_DIM; kt += 64) {
        const int8_t* ap = arow + kt;
        v2i a0 = *reinterpret_cast<const v2i*>(ap);
        v2i a1 = *reinterpret_cast<const v2i*>(ap + 8);
        v2i a2 = *reinterpret_cast<const v2i*>(ap + 32);
        v2i a3 = *reinterpret_cast<const v2i*>(ap + 40);
        v8i a;
        a[0] = a0[0]; a[1] = a0[1];
        a[2] = a1[0]; a[3] = a1[1];
        a[4] = a2[0]; a[5] = a2[1];
        a[6] = a3[0]; a[7] = a3[1];

        v8i b[8];
#pragma unroll
        for (int t = 0; t < 8; ++t) {
            const int8_t* bp = wq + (size_t)(n0 + t * 16 + ln) * K_DIM + kt + hi * 16;
            v4i blo = *reinterpret_cast<const v4i*>(bp);
            v4i bhigh = *reinterpret_cast<const v4i*>(bp + 32);
            b[t][0] = blo[0];  b[t][1] = blo[1];
            b[t][2] = blo[2];  b[t][3] = blo[3];
            b[t][4] = bhigh[0]; b[t][5] = bhigh[1];
            b[t][6] = bhigh[2]; b[t][7] = bhigh[3];
        }

#pragma unroll
        for (int t = 0; t < 8; ++t) {
            acc[t] = __builtin_amdgcn_wmma_i32_16x16x64_iu8(
                 true, a,  true, b[t], acc[t],
                 false,  false);
        }
    }

    float is[8];
#pragma unroll
    for (int v = 0; v < 8; ++v)
        is[v] = iscale[m0 + hi * 8 + v];

#pragma unroll
    for (int t = 0; t < 8; ++t) {
        const int c  = n0 + t * 16 + ln;
        const float ws = wscale[c];
        const float bz = bias[c];
        float* op = out + (size_t)(m0 + hi * 8) * N_DIM + c;
#pragma unroll
        for (int v = 0; v < 8; ++v)
            op[(size_t)v * N_DIM] = (float)acc[t][v] * ws * is[v] + bz;
    }
}

extern "C" void kernel_launch(void* const* d_in, const int* in_sizes, int n_in,
                              void* d_out, int out_size, void* d_ws, size_t ws_size,
                              hipStream_t stream) {
    const float* x      = (const float*)d_in[0];
    const int*   w32    = (const int*)d_in[1];
    const float* wscale = (const float*)d_in[2];
    const float* bias   = (const float*)d_in[3];
    float*       out    = (float*)d_out;

    int8_t* xq     = (int8_t*)d_ws;
    int8_t* w8     = (int8_t*)d_ws + (size_t)M_TOK * K_DIM;
    float*  iscale = (float*)((char*)d_ws + (size_t)M_TOK * K_DIM
                                          + (size_t)N_DIM * K_DIM);

    pack_w_kernel<<<(N_DIM * K_DIM / 4) / 256, 256, 0, stream>>>(w32, (uint32_t*)w8);

    quant_kernel<<<M_TOK / 8, 256, 0, stream>>>(x, (uint32_t*)xq, iscale);

    gemm_kernel<<<dim3(N_DIM / 128, M_TOK / 128), 256, 0, stream>>>(
        xq, iscale, w8, wscale, bias, out);
}
